// HybridQuantumDCNN_59107339927692
// MI455X (gfx1250) — compile-verified, matrix operand order checked by reading, not run on hardware
//
#include <hip/hip_runtime.h>
#include <math.h>
#include <stdint.h>

typedef __attribute__((ext_vector_type(16))) _Float16 v16h;
typedef __attribute__((ext_vector_type(8)))  float    v8f;
typedef __attribute__((ext_vector_type(4)))  unsigned int u32x4;
typedef __attribute__((ext_vector_type(8)))  int      i32x8;
typedef __attribute__((ext_vector_type(4)))  int      i32x4;

#define BVAL 32768
#define FEAT 256
#define HID  64
#define NQ   8
#define MID  32
#define OUTD 10
#define PI_F 3.14159265358979323846f

// ---------------------------------------------------------------------------
// Kernel A: H2 = relu(bn1(relu(x @ w1^T + b1) @ w2^T + b2))
// 256 threads = 8 waves. TDM stages the block's 32x256 f32 x-tile into LDS
// once (kills the 4x global re-read by the column waves), then each wave
// computes a 16x16 H1 tile with 8x v_wmma_f32_16x16x32_f16, then the 64->8
// GEMM + bn1 + relu is fused through a second LDS tile.
// ---------------------------------------------------------------------------
__global__ __launch_bounds__(256) void mlp_front_kernel(
    const float* __restrict__ x,     // (32768,256)
    const float* __restrict__ w1,    // (64,256)
    const float* __restrict__ b1,    // (64,)
    const float* __restrict__ w2,    // (8,64)
    const float* __restrict__ b2,    // (8,)
    const float* __restrict__ bn1g,  // (8,)
    const float* __restrict__ bn1b,  // (8,)
    float* __restrict__ h2out)       // (32768,8)
{
    __shared__ float xtile[32][FEAT];   // 32 KB: staged by TDM
    __shared__ float htile[32][HID];    // 8 KB: H1 tile for fused GEMM2

    const int tid   = threadIdx.x;
    const int lane  = tid & 31;
    const int wave  = tid >> 5;          // 0..7
    const int waveM = wave >> 2;         // 0..1  (16-row slab)
    const int waveN = wave & 3;          // 0..3  (16-col slab)
    const int khi   = (lane >> 4) & 1;   // lane group selects K-half

    // ---- TDM: stage x[blockRow .. blockRow+31][0..255] -> LDS (wave 0) ----
    if (tid < 32) {
        const uint64_t ga  = (uint64_t)(uintptr_t)(x + (size_t)blockIdx.x * 32 * FEAT);
        const uint32_t lds = (uint32_t)(uintptr_t)(&xtile[0][0]);
        // D# group 0: count=1 | lds_addr | global_addr[56:0] | type=2
        u32x4 g0 = { 1u,
                     lds,
                     (uint32_t)ga,
                     (uint32_t)((ga >> 32) & 0x01FFFFFFu) | (2u << 30) };
        // D# group 1: data_size=4B; tensor_dim0=256; tensor_dim1=32768;
        //             tile_dim0=256; tile_dim1=32; tensor_dim0_stride=256
        i32x8 g1 = { (int)0x00020000u,          // data_size=2 (4 bytes)
                     (int)((FEAT & 0xFFFF) << 16),   // tensor_dim0[15:0]
                     (int)0x80000000u,          // tensor_dim1[15:0]=32768
                     (int)(FEAT << 16),         // tile_dim0=256
                     32,                        // tile_dim1=32
                     FEAT,                      // tensor_dim0_stride[31:0]
                     0, 0 };
        i32x4 g2 = { 0, 0, 0, 0 };
        i32x4 g3 = { 0, 0, 0, 0 };
        i32x8 g4 = { 0, 0, 0, 0, 0, 0, 0, 0 };
        __builtin_amdgcn_tensor_load_to_lds(g0, g1, g2, g3, g4, 0);
        __builtin_amdgcn_s_wait_tensorcnt(0);
    }
    __syncthreads();

    const int lrow = waveM * 16 + (lane & 15);   // local row in x-tile (0..31)
    const int colBase = waveN * 16;
    const int col = colBase + (lane & 15);       // B/w1 column index

    v8f acc = {0.f, 0.f, 0.f, 0.f, 0.f, 0.f, 0.f, 0.f};

    // K loop: 8 static WMMAs of 16x16x32 (f16 in, f32 acc)
#pragma unroll
    for (int kb = 0; kb < FEAT; kb += 32) {
        v16h a, b;
#pragma unroll
        for (int e = 0; e < 16; ++e) {
            // ISA 16-bit A/B fragment layout:
            // lanes 0-15: K = e (e<8) or e+8 ; lanes 16-31: shifted by +8
            const int k = kb + e + ((e >= 8) ? 8 : 0) + (khi ? 8 : 0);
            a[e] = (_Float16)xtile[lrow][k];
            b[e] = (_Float16)w1[col * FEAT + k];
        }
        acc = __builtin_amdgcn_wmma_f32_16x16x32_f16(
            /*neg_a=*/false, a, /*neg_b=*/false, b,
            /*c_mod=*/(short)0, acc, /*reuse_a=*/false, /*reuse_b=*/false);
    }

    // Epilogue: bias + relu -> LDS tile (C layout: vgpr r -> M=r(+8), N=lane&15)
    const float bias = b1[col];
#pragma unroll
    for (int r = 0; r < 8; ++r) {
        const int m = r + (khi ? 8 : 0);
        float v = acc[r] + bias;
        v = fmaxf(v, 0.0f);
        htile[waveM * 16 + m][colBase + (lane & 15)] = v;
    }
    __syncthreads();

    // Fused 64 -> 8 GEMM + bn1 + relu. One thread per output element.
    const int rl = tid >> 3;   // 0..31 local row
    const int o  = tid & 7;    // 0..7 output feature
    float s = b2[o];
#pragma unroll 8
    for (int c = 0; c < HID; ++c)
        s += htile[rl][c] * w2[o * HID + c];
    s = s * (bn1g[o] * rsqrtf(1.0f + 1e-5f)) + bn1b[o];
    s = fmaxf(s, 0.0f);
    h2out[(blockIdx.x * 32 + rl) * NQ + o] = s;
}

// ---------------------------------------------------------------------------
// Quantum circuit helpers: one sample per wave32.
// Amplitude index a[7:0] = (reg[2:0] << 5) | lane[4:0].
// Qubits 0-4 -> lane bits (shfl_xor), qubits 5-7 -> register bits.
// ---------------------------------------------------------------------------
__device__ __forceinline__ void ry_gate(float (&re)[8], float (&im)[8],
                                        int q, float th, int lane) {
    float s, c;
    __sincosf(0.5f * th, &s, &c);
    if (q < 5) {
        const int m  = 1 << q;
        const bool hi = (lane >> q) & 1;
#pragma unroll
        for (int k = 0; k < 8; ++k) {
            const float pr = __shfl_xor(re[k], m, 32);
            const float pi = __shfl_xor(im[k], m, 32);
            re[k] = hi ? (s * pr + c * re[k]) : (c * re[k] - s * pr);
            im[k] = hi ? (s * pi + c * im[k]) : (c * im[k] - s * pi);
        }
    } else {
        const int b = 1 << (q - 5);
#pragma unroll
        for (int k = 0; k < 8; ++k) {
            if (!(k & b)) {
                const int k1 = k | b;
                const float r0 = re[k], i0 = im[k], r1 = re[k1], i1 = im[k1];
                re[k]  = c * r0 - s * r1;  im[k]  = c * i0 - s * i1;
                re[k1] = s * r0 + c * r1;  im[k1] = s * i0 + c * i1;
            }
        }
    }
}

__device__ __forceinline__ void phase_gate(float (&re)[8], float (&im)[8],
                                           int q, float phi, int lane) {
    float sp, cp;
    __sincosf(phi, &sp, &cp);
    if (q < 5) {
        const bool hi = (lane >> q) & 1;
#pragma unroll
        for (int k = 0; k < 8; ++k) {
            const float r = re[k], i = im[k];
            const float nr = r * cp - i * sp;
            const float ni = r * sp + i * cp;
            re[k] = hi ? nr : r;
            im[k] = hi ? ni : i;
        }
    } else {
        const int b = 1 << (q - 5);
#pragma unroll
        for (int k = 0; k < 8; ++k) {
            if (k & b) {
                const float r = re[k], i = im[k];
                re[k] = r * cp - i * sp;
                im[k] = r * sp + i * cp;
            }
        }
    }
}

__device__ __forceinline__ void cnot_gate(float (&re)[8], float (&im)[8],
                                          int c, int t, int lane) {
    if (c < 5 && t < 5) {
        const int m = 1 << t;
        const bool ctrl = (lane >> c) & 1;
#pragma unroll
        for (int k = 0; k < 8; ++k) {
            const float pr = __shfl_xor(re[k], m, 32);
            const float pi = __shfl_xor(im[k], m, 32);
            re[k] = ctrl ? pr : re[k];
            im[k] = ctrl ? pi : im[k];
        }
    } else if (c < 5 && t >= 5) {
        const int b = 1 << (t - 5);
        const bool ctrl = (lane >> c) & 1;
#pragma unroll
        for (int k = 0; k < 8; ++k) {
            if (!(k & b)) {
                const int k1 = k | b;
                const float r0 = re[k], i0 = im[k], r1 = re[k1], i1 = im[k1];
                re[k]  = ctrl ? r1 : r0;  im[k]  = ctrl ? i1 : i0;
                re[k1] = ctrl ? r0 : r1;  im[k1] = ctrl ? i0 : i1;
            }
        }
    } else if (c >= 5 && t < 5) {
        const int cb = 1 << (c - 5);
        const int m  = 1 << t;
#pragma unroll
        for (int k = 0; k < 8; ++k) {
            if (k & cb) {
                re[k] = __shfl_xor(re[k], m, 32);
                im[k] = __shfl_xor(im[k], m, 32);
            }
        }
    } else {
        const int cb = 1 << (c - 5);
        const int tb = 1 << (t - 5);
#pragma unroll
        for (int k = 0; k < 8; ++k) {
            if ((k & cb) && !(k & tb)) {
                const int k1 = k | tb;
                float tr = re[k]; re[k] = re[k1]; re[k1] = tr;
                float ti = im[k]; im[k] = im[k1]; im[k1] = ti;
            }
        }
    }
}

__device__ __forceinline__ float wave_sum(float v) {
#pragma unroll
    for (int o = 16; o > 0; o >>= 1)
        v += __shfl_xor(v, o, 32);
    return v;
}

// ---------------------------------------------------------------------------
// Kernel B: circuit + tail MLP. One sample per wave; 8 waves per block.
// ---------------------------------------------------------------------------
__global__ __launch_bounds__(256) void quantum_tail_kernel(
    const float* __restrict__ h2,    // (32768,8) from workspace
    const float* __restrict__ qw,    // (48,)
    const float* __restrict__ w3,    // (32,8)
    const float* __restrict__ b3,    // (32,)
    const float* __restrict__ bn2g,  // (32,)
    const float* __restrict__ bn2b,  // (32,)
    const float* __restrict__ w4,    // (10,32)
    const float* __restrict__ b4,    // (10,)
    float* __restrict__ out)         // (32768,10)
{
    const int lane   = threadIdx.x & 31;
    const int wave   = threadIdx.x >> 5;
    const int sample = blockIdx.x * 8 + wave;
    if (sample >= BVAL) return;

    const float* hp = h2 + sample * NQ;

    // |00000000>
    float re[8], im[8];
#pragma unroll
    for (int k = 0; k < 8; ++k) { re[k] = 0.0f; im[k] = 0.0f; }
    if (lane == 0) re[0] = 1.0f;

    // Encoding RYs
#pragma unroll
    for (int q = 0; q < NQ; ++q)
        ry_gate(re, im, q, hp[q] * PI_F, lane);

    // Entangling + variational layers
#pragma unroll
    for (int layer = 0; layer < 3; ++layer) {
        const int dil = 1 << layer;
        const float* rw = qw + layer * (2 * NQ);
        const float* pw = rw + NQ;
        for (int i = 0; i < NQ - dil; ++i)
            cnot_gate(re, im, i, i + dil, lane);
#pragma unroll
        for (int i = 0; i < NQ; ++i) {
            const float pf = __sinf((float)(i + 1) * (PI_F / (float)NQ));
            ry_gate(re, im, i, rw[i] * pf, lane);
            phase_gate(re, im, i, pw[i] * pf, lane);
        }
    }

    // Measurement: z[q] = <Z_q> = sum prob * (+1/-1)
    float P = 0.0f, S5 = 0.0f, S6 = 0.0f, S7 = 0.0f;
#pragma unroll
    for (int k = 0; k < 8; ++k) {
        const float pr = re[k] * re[k] + im[k] * im[k];
        P  += pr;
        S5 += ((k >> 0) & 1) ? -pr : pr;
        S6 += ((k >> 1) & 1) ? -pr : pr;
        S7 += ((k >> 2) & 1) ? -pr : pr;
    }
    float z[8];
#pragma unroll
    for (int q = 0; q < 5; ++q)
        z[q] = wave_sum(((lane >> q) & 1) ? -P : P);
    z[5] = wave_sum(S5);
    z[6] = wave_sum(S6);
    z[7] = wave_sum(S7);

    // p = relu(bn2(z @ w3^T + b3)); 32 lanes = 32 hidden units
    float p = b3[lane];
#pragma unroll
    for (int j = 0; j < NQ; ++j)
        p += z[j] * w3[lane * NQ + j];
    p = p * (bn2g[lane] * rsqrtf(1.0f + 1e-5f)) + bn2b[lane];
    p = fmaxf(p, 0.0f);

    // out = p @ w4^T + b4 via 10 shuffle reductions
#pragma unroll
    for (int o = 0; o < OUTD; ++o) {
        float v = wave_sum(p * w4[o * MID + lane]);
        if (lane == 0)
            out[sample * OUTD + o] = v + b4[o];
    }
}

// ---------------------------------------------------------------------------
extern "C" void kernel_launch(void* const* d_in, const int* in_sizes, int n_in,
                              void* d_out, int out_size, void* d_ws, size_t ws_size,
                              hipStream_t stream) {
    const float* x     = (const float*)d_in[0];
    const float* w1    = (const float*)d_in[1];
    const float* b1    = (const float*)d_in[2];
    const float* w2    = (const float*)d_in[3];
    const float* b2    = (const float*)d_in[4];
    const float* bn1g  = (const float*)d_in[5];
    const float* bn1b  = (const float*)d_in[6];
    const float* qw    = (const float*)d_in[7];
    const float* w3    = (const float*)d_in[8];
    const float* b3    = (const float*)d_in[9];
    const float* bn2g  = (const float*)d_in[10];
    const float* bn2b  = (const float*)d_in[11];
    const float* w4    = (const float*)d_in[12];
    const float* b4    = (const float*)d_in[13];
    float* out = (float*)d_out;
    float* h2  = (float*)d_ws;   // 32768*8 floats = 1 MB scratch

    mlp_front_kernel<<<BVAL / 32, 256, 0, stream>>>(
        x, w1, b1, w2, b2, bn1g, bn1b, h2);

    quantum_tail_kernel<<<BVAL / 8, 256, 0, stream>>>(
        h2, qw, w3, b3, bn2g, bn2b, w4, b4, out);
}
